// GraphNetwork_22986664968693
// MI455X (gfx1250) — matrix operand order repaired and checked by reading, not compiled, not run on hardware
//
#include <hip/hip_runtime.h>

typedef __bf16 bf16_t;
typedef __attribute__((ext_vector_type(16))) __bf16 v16bf;
typedef __attribute__((ext_vector_type(8)))  __bf16 v8bf;
typedef __attribute__((ext_vector_type(8)))  float  v8f;

typedef unsigned int v4u __attribute__((ext_vector_type(4)));
typedef int          v8i __attribute__((ext_vector_type(8)));
typedef int          v4i __attribute__((ext_vector_type(4)));

#if defined(__has_builtin)
#  if __has_builtin(__builtin_amdgcn_tensor_load_to_lds) && \
      __has_builtin(__builtin_amdgcn_s_wait_tensorcnt)
#    define USE_TDM 1
#  endif
#endif

__device__ __forceinline__ v8f zero_v8f() {
  v8f z;
#pragma unroll
  for (int i = 0; i < 8; ++i) z[i] = 0.0f;
  return z;
}

// ---------------- utility kernels ----------------

__global__ void zero_f32_kernel(float* __restrict__ p, long long n) {
  long long i = (long long)blockIdx.x * blockDim.x + threadIdx.x;
  if (i < n) p[i] = 0.0f;
}

__global__ void cvt_f32_to_bf16_kernel(const float* __restrict__ in,
                                       bf16_t* __restrict__ out, long long n) {
  long long i = (long long)blockIdx.x * blockDim.x + threadIdx.x;
  if (i < n) out[i] = (bf16_t)in[i];
}

__global__ void relu_cvt_kernel(const float* __restrict__ agg,
                                bf16_t* __restrict__ h, long long n) {
  long long i = (long long)blockIdx.x * blockDim.x + threadIdx.x;
  if (i < n) h[i] = (bf16_t)fmaxf(agg[i], 0.0f);
}

// Fourier basis ('ffourier', TERMS=4, DIM=2, mode='cat'):
// per dim: [1, sin(pi x), cos(pi x), sin(2 pi x)] -> 8 bf16 per edge.
__global__ void basis_kernel(const float* __restrict__ el,
                             bf16_t* __restrict__ basis, int E) {
  int e = blockIdx.x * blockDim.x + threadIdx.x;
  if (e >= E) return;
  const float PI = 3.14159265358979f;
  v8bf b;
#pragma unroll
  for (int d = 0; d < 2; ++d) {
    float x = el[(long long)e * 2 + d];
    b[d * 4 + 0] = (bf16_t)1.0f;
    b[d * 4 + 1] = (bf16_t)__sinf(PI * x);
    b[d * 4 + 2] = (bf16_t)__cosf(PI * x);
    b[d * 4 + 3] = (bf16_t)__sinf(2.0f * PI * x);
  }
  *(v8bf*)(basis + (long long)e * 8) = b;
}

// Repack W [K=8*IN_F, OUT_F] (f32, row-major; k = b*IN_F + f) into WMMA
// B-fragment order (bf16): frag = c*NT + nt; within frag: [lane][j] with
// element j of lane L -> k = 32*c + 8*(L>>4) + (j&7) + 16*(j>>3), n = (L&15) + 16*nt.
template<int IN_F, int OUT_F, int NT>
__global__ void repack_w_kernel(const float* __restrict__ W,
                                bf16_t* __restrict__ Wfrag) {
  constexpr int NCHUNK = (8 * IN_F) / 32;
  constexpr int TOTAL  = NCHUNK * NT * 512;
  int idx = blockIdx.x * blockDim.x + threadIdx.x;
  if (idx >= TOTAL) return;
  int j    = idx & 15;
  int lane = (idx >> 4) & 31;
  int frag = idx >> 9;
  int nt = frag % NT;
  int c  = frag / NT;
  int h  = lane >> 4;
  int n  = (lane & 15) + 16 * nt;
  int k  = 32 * c + 8 * h + (j & 7) + 16 * (j >> 3);
  float v = (n < OUT_F) ? W[(long long)k * OUT_F + n] : 0.0f;
  Wfrag[idx] = (bf16_t)v;
}

// Scalar fallback for E % 16 leftover edges (rem == 0 for this problem size).
__global__ void edge_tail_kernel(const bf16_t* __restrict__ hfeat,
                                 const bf16_t* __restrict__ basis,
                                 const int* __restrict__ src,
                                 const int* __restrict__ dst,
                                 const float* __restrict__ W,
                                 float* __restrict__ agg,
                                 int E, int rem, int IN_F, int OUT_F,
                                 float scale) {
  int idx = blockIdx.x * blockDim.x + threadIdx.x;
  if (idx >= rem * OUT_F) return;
  int e = E - rem + idx / OUT_F;
  int o = idx % OUT_F;
  int s = src[e], d = dst[e];
  float sum = 0.0f;
  for (int b = 0; b < 8; ++b) {
    float bv = (float)basis[(long long)e * 8 + b];
    for (int f = 0; f < IN_F; ++f)
      sum += bv * (float)hfeat[(long long)s * IN_F + f] *
             W[(long long)(b * IN_F + f) * OUT_F + o];
  }
  unsafeAtomicAdd(&agg[(long long)d * OUT_F + o], sum * scale);
}

// ---------------- main WMMA edge pass ----------------
// One wave per 16-edge tile. K = 8*IN_F contracted in chunks of 32 with
// v_wmma_f32_16x16x32_bf16; A built on the fly as basis[b] * feat[f]
// (lowered to v_pk_mul_bf16); B fragments streamed from LDS (filled via the
// Tensor Data Mover when available); C scattered with f32 global atomics.
template<int IN_F, int OUT_F, int NT, int LOG2F>
__global__ __launch_bounds__(256)
void edge_pass_kernel(const bf16_t* __restrict__ hfeat,
                      const bf16_t* __restrict__ basis,
                      const int* __restrict__ src,
                      const int* __restrict__ dst,
                      const bf16_t* __restrict__ Wfrag,
                      float* __restrict__ agg,
                      int E, float scale) {
  constexpr int NCHUNK = (8 * IN_F) / 32;
  constexpr int FRAGS  = NCHUNK * NT;
  constexpr int LDSN   = FRAGS * 512;           // bf16 elements
  static_assert(IN_F >= 16, "IN_F must be >= 16");
  __shared__ __align__(16) bf16_t ldsW[LDSN];

#ifdef USE_TDM
  // TDM bulk copy: one 1-D tile of 8-byte elements, issued by wave 0 only
  // (TDM ignores EXEC and issues per-wave; one DMA per workgroup suffices).
  if (threadIdx.x < 32) {
    constexpr unsigned nelem = (unsigned)(LDSN / 4);   // 8-byte elements
    unsigned lds_off = (unsigned)(size_t)(void*)&ldsW[0]; // low 32b of generic LDS addr = LDS offset
    unsigned long long ga = (unsigned long long)(const void*)Wfrag;
    v4u g0;
    g0[0] = 1u;                                        // count=1, user descriptor
    g0[1] = lds_off;                                   // lds_addr
    g0[2] = (unsigned)(ga & 0xFFFFFFFFu);              // global_addr[31:0]
    g0[3] = (unsigned)((ga >> 32) & 0x1FFFFFFu) | (2u << 30);  // addr[56:32], type=2
    v8i g1;
    g1[0] = (int)(3u << 16);                           // data_size=3 (8B), no flags
    g1[1] = (int)((nelem & 0xFFFFu) << 16);            // tensor_dim0[15:0]
    g1[2] = (int)(((nelem >> 16) & 0xFFFFu) | (1u << 16)); // tensor_dim0[31:16], tensor_dim1=1
    g1[3] = (int)(nelem << 16);                        // tile_dim0 = nelem (<= 8192)
    g1[4] = 1;                                         // tile_dim1=1, tile_dim2=0
    g1[5] = (int)nelem;                                // tensor_dim0_stride[31:0]
    g1[6] = 0;                                         // stride hi, dim1_stride lo
    g1[7] = 0;
    v4i gz = {0, 0, 0, 0};
#if defined(__clang_major__) && __clang_major__ >= 23
    v8i gz8 = {0, 0, 0, 0, 0, 0, 0, 0};
    __builtin_amdgcn_tensor_load_to_lds(g0, g1, gz, gz, gz8, 0);
#else
    __builtin_amdgcn_tensor_load_to_lds(g0, g1, gz, gz, 0);
#endif
    __builtin_amdgcn_s_wait_tensorcnt(0);
  }
  __syncthreads();
#else
  {
    const uint4* g = (const uint4*)Wfrag;
    uint4* l = (uint4*)ldsW;
#pragma unroll 1
    for (int i = threadIdx.x; i < LDSN / 8; i += blockDim.x) l[i] = g[i];
  }
  __syncthreads();
#endif

  const int lane = threadIdx.x & 31;
  const int r = lane & 15;
  const int h = lane >> 4;
  const int wid = (int)((blockIdx.x * blockDim.x + threadIdx.x) >> 5);
  const int nw  = (int)((gridDim.x * blockDim.x) >> 5);
  const int ntiles = E >> 4;
  constexpr int NQ = (IN_F >= 64) ? 4 : 1;

  for (int tile = wid; tile < ntiles; tile += nw) {
    const int ebase = tile << 4;
    const int s = src[ebase + r];
    const int d = dst[ebase + r];
    v8bf brow = *(const v8bf*)(basis + (long long)(ebase + r) * 8);
    v8bf fv[NQ];
#pragma unroll
    for (int q = 0; q < NQ; ++q)
      fv[q] = *(const v8bf*)(hfeat + (long long)s * IN_F + 16 * q + 8 * h);

    v8f acc[NT];
#pragma unroll
    for (int nt = 0; nt < NT; ++nt) acc[nt] = zero_v8f();

#pragma unroll
    for (int c = 0; c < NCHUNK; ++c) {
      // A element j (ISA 16-bit A layout, lane half h):
      //   j in [0,8):  k = 32c + 8h + j
      //   j in [8,16): k = 32c + 16 + 8h + (j-8)
      // b = k >> LOG2F and feat run selection are h-invariant (8h < 16).
      const int b0 = (32 * c) >> LOG2F;
      const int b1 = (32 * c + 16) >> LOG2F;
      const int q0 = ((32 * c) & (IN_F - 1)) >> 4;
      const int q1 = ((32 * c + 16) & (IN_F - 1)) >> 4;
      const float s0 = (float)brow[b0];
      const float s1 = (float)brow[b1];
      v16bf a;
#pragma unroll
      for (int j = 0; j < 8; ++j) a[j] = (bf16_t)(s0 * (float)fv[q0][j]);
#pragma unroll
      for (int j = 0; j < 8; ++j) a[8 + j] = (bf16_t)(s1 * (float)fv[q1][j]);

#pragma unroll
      for (int nt = 0; nt < NT; ++nt) {
        v16bf bfrag = *(const v16bf*)&ldsW[((c * NT + nt) * 32 + lane) * 16];
        acc[nt] = __builtin_amdgcn_wmma_f32_16x16x32_bf16(
            false, a, false, bfrag, (short)0, acc[nt], false, false);
      }
    }

    // Scatter: C/D layout: VGPR rr holds row M = rr + 8h, col N = (lane&15)+16nt.
#pragma unroll
    for (int rr = 0; rr < 8; ++rr) {
      const int m = rr + 8 * h;
      const int dd = __builtin_amdgcn_ds_bpermute(m << 2, d);
#pragma unroll
      for (int nt = 0; nt < NT; ++nt) {
        const int n = (lane & 15) + 16 * nt;
        if (n < OUT_F) {
          unsafeAtomicAdd(&agg[(long long)dd * OUT_F + n], acc[nt][rr] * scale);
        }
      }
    }
  }
}

// ---------------- host-side launcher ----------------

extern "C" void kernel_launch(void* const* d_in, const int* in_sizes, int n_in,
                              void* d_out, int out_size, void* d_ws, size_t ws_size,
                              hipStream_t stream) {
  const float* x  = (const float*)d_in[0];
  const int*   ei = (const int*)d_in[1];
  const float* el = (const float*)d_in[2];
  const float* W0 = (const float*)d_in[3];
  const float* W1 = (const float*)d_in[4];
  const float* W2 = (const float*)d_in[5];
  const float* W3 = (const float*)d_in[6];

  const int Nn = in_sizes[0] / 16;   // nodes
  const int E  = in_sizes[1] / 2;    // edges
  const int* src = ei;
  const int* dst = ei + E;

  // workspace carving (256B aligned)
  char* ws = (char*)d_ws;
  size_t off = 0;
  auto carve = [&](size_t bytes) -> char* {
    char* p = ws + off;
    off = (off + bytes + 255) & ~(size_t)255;
    return p;
  };
  bf16_t* basis = (bf16_t*)carve((size_t)E * 8 * sizeof(bf16_t));
  bf16_t* h0    = (bf16_t*)carve((size_t)Nn * 16 * sizeof(bf16_t));
  bf16_t* h     = (bf16_t*)carve((size_t)Nn * 64 * sizeof(bf16_t));
  float*  agg   = (float*)carve((size_t)Nn * 64 * sizeof(float));
  bf16_t* Wf0   = (bf16_t*)carve((size_t)4  * 4 * 512 * sizeof(bf16_t));
  bf16_t* Wf1   = (bf16_t*)carve((size_t)16 * 4 * 512 * sizeof(bf16_t));
  bf16_t* Wf2   = (bf16_t*)carve((size_t)16 * 4 * 512 * sizeof(bf16_t));
  bf16_t* Wf3   = (bf16_t*)carve((size_t)16 * 1 * 512 * sizeof(bf16_t));

  const int T = 256;
  const int ntiles = E >> 4;
  const int rem = E & 15;
  int eblocks = ntiles / 8 + 1;
  if (eblocks > 2048) eblocks = 2048;
  float* outp = (float*)d_out;

  // ---- prep ----
  basis_kernel<<<(E + T - 1) / T, T, 0, stream>>>(el, basis, E);
  cvt_f32_to_bf16_kernel<<<((long long)Nn * 16 + T - 1) / T, T, 0, stream>>>(x, h0, (long long)Nn * 16);
  repack_w_kernel<16, 64, 4><<<(4  * 4 * 512 + T - 1) / T, T, 0, stream>>>(W0, Wf0);
  repack_w_kernel<64, 64, 4><<<(16 * 4 * 512 + T - 1) / T, T, 0, stream>>>(W1, Wf1);
  repack_w_kernel<64, 64, 4><<<(16 * 4 * 512 + T - 1) / T, T, 0, stream>>>(W2, Wf2);
  repack_w_kernel<64,  2, 1><<<(16 * 1 * 512 + T - 1) / T, T, 0, stream>>>(W3, Wf3);

  // ---- layer 0: 16 -> 64 ----
  zero_f32_kernel<<<((long long)Nn * 64 + T - 1) / T, T, 0, stream>>>(agg, (long long)Nn * 64);
  edge_pass_kernel<16, 64, 4, 4><<<eblocks, T, 0, stream>>>(h0, basis, src, dst, Wf0, agg, E, 1.0f);
  if (rem)
    edge_tail_kernel<<<(rem * 64 + T - 1) / T, T, 0, stream>>>(h0, basis, src, dst, W0, agg, E, rem, 16, 64, 1.0f);
  relu_cvt_kernel<<<((long long)Nn * 64 + T - 1) / T, T, 0, stream>>>(agg, h, (long long)Nn * 64);

  // ---- layer 1: 64 -> 64 ----
  zero_f32_kernel<<<((long long)Nn * 64 + T - 1) / T, T, 0, stream>>>(agg, (long long)Nn * 64);
  edge_pass_kernel<64, 64, 4, 6><<<eblocks, T, 0, stream>>>(h, basis, src, dst, Wf1, agg, E, 1.0f);
  if (rem)
    edge_tail_kernel<<<(rem * 64 + T - 1) / T, T, 0, stream>>>(h, basis, src, dst, W1, agg, E, rem, 64, 64, 1.0f);
  relu_cvt_kernel<<<((long long)Nn * 64 + T - 1) / T, T, 0, stream>>>(agg, h, (long long)Nn * 64);

  // ---- layer 2: 64 -> 64 ----
  zero_f32_kernel<<<((long long)Nn * 64 + T - 1) / T, T, 0, stream>>>(agg, (long long)Nn * 64);
  edge_pass_kernel<64, 64, 4, 6><<<eblocks, T, 0, stream>>>(h, basis, src, dst, Wf2, agg, E, 1.0f);
  if (rem)
    edge_tail_kernel<<<(rem * 64 + T - 1) / T, T, 0, stream>>>(h, basis, src, dst, W2, agg, E, rem, 64, 64, 1.0f);
  relu_cvt_kernel<<<((long long)Nn * 64 + T - 1) / T, T, 0, stream>>>(agg, h, (long long)Nn * 64);

  // ---- layer 3: 64 -> 2 (no relu, scaled 1/128, direct to d_out) ----
  zero_f32_kernel<<<((long long)Nn * 2 + T - 1) / T, T, 0, stream>>>(outp, (long long)Nn * 2);
  edge_pass_kernel<64, 2, 1, 6><<<eblocks, T, 0, stream>>>(h, basis, src, dst, Wf3, outp, E, 1.0f / 128.0f);
  if (rem)
    edge_tail_kernel<<<(rem * 2 + T - 1) / T, T, 0, stream>>>(h, basis, src, dst, W3, outp, E, rem, 64, 2, 1.0f / 128.0f);
}
